// stack_latent_attention_66554813219293
// MI455X (gfx1250) — matrix operand order repaired and checked by reading, not compiled, not run on hardware
//
#include <hip/hip_runtime.h>
#include <stdint.h>

// Problem constants (match reference)
#define B_   16
#define NV_  1024
#define NH_  1024
#define D_   1024
#define NEGBIG (-1e30f)

typedef __attribute__((ext_vector_type(16))) __bf16        bf16x16;
typedef __attribute__((ext_vector_type(8)))  float         f32x8;
typedef __attribute__((ext_vector_type(4)))  unsigned int  u32x4;

union FragU { u32x4 q[2]; bf16x16 v; };

__device__ __forceinline__ unsigned short f2bf(float f) {
    union { float f; unsigned int u; } c; c.f = f;
    unsigned int x = c.u;
    unsigned int r = x + 0x7fffu + ((x >> 16) & 1u);   // round-to-nearest-even
    return (unsigned short)(r >> 16);
}

// CDNA5 async global->LDS copy (16B per lane), tracked by ASYNCcnt.
__device__ __forceinline__ void async_b128(unsigned lds_off, const void* g) {
    asm volatile("global_load_async_to_lds_b128 %0, %1, off"
                 :: "v"(lds_off), "v"((unsigned long long)(uintptr_t)g)
                 : "memory");
}
__device__ __forceinline__ void wait_async0() {
    asm volatile("s_wait_asynccnt 0" ::: "memory");
}
__device__ __forceinline__ unsigned lds_off(const void* p) {
    return (unsigned)(uintptr_t)p;   // low 32 bits of generic LDS address = offset
}

// ---------------------------------------------------------------------------
// f32 -> bf16 elementwise conversion (grid-stride)
// ---------------------------------------------------------------------------
__global__ void k_f32_to_bf16(const float* __restrict__ src,
                              unsigned short* __restrict__ dst, size_t n) {
    size_t i = (size_t)blockIdx.x * blockDim.x + threadIdx.x;
    size_t stride = (size_t)gridDim.x * blockDim.x;
    for (; i < n; i += stride) dst[i] = f2bf(src[i]);
}

// ---------------------------------------------------------------------------
// bf16 WMMA GEMM:  C[M,N] (+)= A[M,K] * B
//   BL==0 (NT): B given as [N,K] row-major (torch Linear weight / q.k^T)
//   BL==1 (NN): B given as [K,N] row-major
// 256 threads = 8 waves (2x4); WG tile 128x256; wave tile 64x64 (4x4 WMMA).
// K-step 32. Double-buffered LDS staged with global_load_async_to_lds_b128:
// 16 ds_load_b128 feed 16 WMMAs per wave per K-step (512 B/WMMA of LDS reads).
// ---------------------------------------------------------------------------
__device__ __forceinline__ bf16x16 load_a_frag(const unsigned short* lds, int mrow, int lane) {
    // 16-bit A 16x32 layout: lanes 0-15 M=lane, VGPR0-3=K0..7(+8*khalf via lane>>4),
    // VGPR4-7=K16..23(+8*khalf)
    const int khalf = lane >> 4;
    const int row = mrow + (lane & 15);
    const char* base = (const char*)(lds + row * 32);
    FragU f;
    f.q[0] = *(const u32x4*)(base + khalf * 16);
    f.q[1] = *(const u32x4*)(base + 32 + khalf * 16);
    return f.v;
}
__device__ __forceinline__ bf16x16 load_b_frag(const unsigned short* lds, int ncol, int lane) {
    // B 32x16 layout: lanes 0-15 N=lane, VGPR0-3=K0..7(+8*lh), VGPR4-7=K16..23(+8*lh).
    // LDS B tile is stored column(N)-major: [n][k], 32 K per column.
    const int lh = lane >> 4;
    const int col = ncol + (lane & 15);
    const char* base = (const char*)(lds + col * 32);
    FragU f;
    f.q[0] = *(const u32x4*)(base + lh * 16);
    f.q[1] = *(const u32x4*)(base + 32 + lh * 16);
    return f.v;
}

template <int BL, bool BIAS, bool ACCUM, bool OUTBF>
__global__ __launch_bounds__(256)
void k_gemm_bf16(const unsigned short* __restrict__ A, long long sA,
                 const unsigned short* __restrict__ Bm, long long sB,
                 const float* __restrict__ bias,
                 float* __restrict__ Cf, unsigned short* __restrict__ Cb, long long sC,
                 int M, int N, int K, int lda, int ldb, int ldc) {
    __shared__ __attribute__((aligned(16))) unsigned short lsA[2][128 * 32];   // [m][k]
    __shared__ __attribute__((aligned(16))) unsigned short lsB[2][256 * 32];   // [n][k]

    const int tid  = threadIdx.x;
    const int lane = tid & 31;
    const int wave = tid >> 5;
    const int wm   = wave >> 2;   // 0..1 -> 64-row slab
    const int wn   = wave & 3;    // 0..3 -> 64-col slab
    const int m0   = blockIdx.y * 128;
    const int n0   = blockIdx.x * 256;
    const int b    = blockIdx.z;

    const unsigned short* Ab = A  + (size_t)b * sA;
    const unsigned short* Bb = Bm + (size_t)b * sB;

    f32x8 acc[4][4];
#pragma unroll
    for (int mi = 0; mi < 4; ++mi)
#pragma unroll
        for (int ni = 0; ni < 4; ++ni) acc[mi][ni] = (f32x8){};

    // ---- staging helpers ----
    // A tile 128x32: 512 chunks of 8 bf16 -> 2 async copies/thread
    // B tile 256x32 (NT): 1024 chunks -> 4 async copies/thread
    // B tile (NN): 4x u32x4 register fetch + LDS transpose scatter (pipelined)
    u32x4 bregN[4];

    auto stage_a = [&](int k0, int buf) {
#pragma unroll
        for (int i = 0; i < 2; ++i) {
            int c = tid + i * 256;
            int row = c >> 2;
            int col8 = (c & 3) * 8;
            async_b128(lds_off(&lsA[buf][row * 32 + col8]),
                       &Ab[(size_t)(m0 + row) * lda + k0 + col8]);
        }
    };
    auto stage_b_nt = [&](int k0, int buf) {
#pragma unroll
        for (int i = 0; i < 4; ++i) {
            int c = tid + i * 256;
            int row = c >> 2;           // 0..255
            int col8 = (c & 3) * 8;
            async_b128(lds_off(&lsB[buf][row * 32 + col8]),
                       &Bb[(size_t)(n0 + row) * ldb + k0 + col8]);
        }
    };
    auto fetch_b_nn = [&](int k0) {
#pragma unroll
        for (int i = 0; i < 4; ++i) {
            int c = tid + i * 256;
            int kk = c >> 5;            // 0..31
            int n8 = (c & 31) * 8;      // 0..248
            bregN[i] = *(const u32x4*)(&Bb[(size_t)(k0 + kk) * ldb + n0 + n8]);
        }
    };
    auto commit_b_nn = [&](int buf) {
#pragma unroll
        for (int i = 0; i < 4; ++i) {
            int c = tid + i * 256;
            int kk = c >> 5;
            int n8 = (c & 31) * 8;
#pragma unroll
            for (int j = 0; j < 8; ++j) {
                unsigned int w = bregN[i][j >> 1];
                lsB[buf][(size_t)(n8 + j) * 32 + kk] =
                    (unsigned short)((w >> ((j & 1) * 16)) & 0xffffu);
            }
        }
    };

    // ---- prologue: fill buffer 0 ----
    int buf = 0;
    stage_a(0, 0);
    if (BL == 0) {
        stage_b_nt(0, 0);
    } else {
        fetch_b_nn(0);
        commit_b_nn(0);
    }
    wait_async0();
    __syncthreads();

    // ---- main pipelined loop ----
    for (int k0 = 0; k0 < K; k0 += 32) {
        const bool has_next = (k0 + 32 < K);
        if (has_next) {
            stage_a(k0 + 32, buf ^ 1);                 // async, overlaps compute
            if (BL == 0) stage_b_nt(k0 + 32, buf ^ 1); // async
            else         fetch_b_nn(k0 + 32);          // regs, committed after compute
        }
        if (k0 + 64 < K) {  // deep prefetch into L2 (global_prefetch_b8)
            __builtin_prefetch(&Ab[(size_t)(m0 + (tid >> 2)) * lda + k0 + 64 + (tid & 3) * 8], 0, 1);
        }

        bf16x16 af[4];
#pragma unroll
        for (int mi = 0; mi < 4; ++mi)
            af[mi] = load_a_frag(&lsA[buf][0], wm * 64 + mi * 16, lane);
        bf16x16 bfr[4];
#pragma unroll
        for (int ni = 0; ni < 4; ++ni)
            bfr[ni] = load_b_frag(&lsB[buf][0], wn * 64 + ni * 16, lane);
#pragma unroll
        for (int mi = 0; mi < 4; ++mi) {
#pragma unroll
            for (int ni = 0; ni < 4; ++ni) {
                acc[mi][ni] = __builtin_amdgcn_wmma_f32_16x16x32_bf16(
                    false, af[mi], false, bfr[ni], (short)0, acc[mi][ni], false, false);
            }
        }

        if (BL == 1 && has_next) commit_b_nn(buf ^ 1);
        wait_async0();       // next buffer's async copies landed in LDS
        __syncthreads();     // everyone done reading buf / writing buf^1
        buf ^= 1;
    }

    // epilogue: C layout VGPR r -> M = r + 8*(lane>>4), N = lane&15
#pragma unroll
    for (int mi = 0; mi < 4; ++mi) {
#pragma unroll
        for (int ni = 0; ni < 4; ++ni) {
            const int mbase = m0 + wm * 64 + mi * 16 + ((lane >> 4) * 8);
            const int n     = n0 + wn * 64 + ni * 16 + (lane & 15);
            float bv = BIAS ? bias[n] : 0.0f;
#pragma unroll
            for (int r = 0; r < 8; ++r) {
                float vv = acc[mi][ni][r] + bv;
                size_t idx = (size_t)b * sC + (size_t)(mbase + r) * ldc + n;
                if (OUTBF) {
                    Cb[idx] = f2bf(vv);
                } else {
                    if (ACCUM) vv += Cf[idx];
                    Cf[idx] = vv;
                }
            }
        }
    }
}

// ---------------------------------------------------------------------------
// Row softmax of att (f32 [B,NV,NH]) with h_mask, writing bf16 probs
// ---------------------------------------------------------------------------
__global__ __launch_bounds__(256)
void k_softmax_att(const float* __restrict__ att, const int* __restrict__ h_mask,
                   unsigned short* __restrict__ probs) {
    __shared__ float red[256];
    const int i = blockIdx.x, b = blockIdx.y, t = threadIdx.x;
    const float* row  = att + ((size_t)b * NV_ + i) * NH_;
    const int*   mask = h_mask + (size_t)b * NH_;
    float l[4];
    float mx = -3.4e38f;
#pragma unroll
    for (int j = 0; j < 4; ++j) {
        int col = t + j * 256;
        float val = (mask[col] == 0) ? NEGBIG : row[col];
        l[j] = val; mx = fmaxf(mx, val);
    }
    red[t] = mx; __syncthreads();
    for (int s = 128; s > 0; s >>= 1) { if (t < s) red[t] = fmaxf(red[t], red[t + s]); __syncthreads(); }
    mx = red[0]; __syncthreads();
    float sum = 0.0f;
#pragma unroll
    for (int j = 0; j < 4; ++j) { l[j] = __expf(l[j] - mx); sum += l[j]; }
    red[t] = sum; __syncthreads();
    for (int s = 128; s > 0; s >>= 1) { if (t < s) red[t] += red[t + s]; __syncthreads(); }
    float inv = 1.0f / red[0];
    unsigned short* orow = probs + ((size_t)b * NV_ + i) * NH_;
#pragma unroll
    for (int j = 0; j < 4; ++j) orow[t + j * 256] = f2bf(l[j] * inv);
}

// ---------------------------------------------------------------------------
// s = sigmoid(s_pre); write s into z_new[:, :D] of d_out; u = s . Wu
// ---------------------------------------------------------------------------
__global__ __launch_bounds__(256)
void k_sigmoid_u(const float* __restrict__ s_pre, const float* __restrict__ Wu,
                 float* __restrict__ z_out, float* __restrict__ u) {
    __shared__ float red[256];
    const int i = blockIdx.x, b = blockIdx.y, t = threadIdx.x;
    const float* row = s_pre + ((size_t)b * NV_ + i) * D_;
    float* zrow = z_out + ((size_t)b * NV_ + i) * (size_t)(2 * D_);
    float part = 0.0f;
#pragma unroll
    for (int j = 0; j < 4; ++j) {
        int d = t + j * 256;
        float sv = 1.0f / (1.0f + __expf(-row[d]));
        zrow[d] = sv;
        part += sv * Wu[d];
    }
    red[t] = part; __syncthreads();
    for (int s = 128; s > 0; s >>= 1) { if (t < s) red[t] += red[t + s]; __syncthreads(); }
    if (t == 0) u[(size_t)b * NV_ + i] = red[0];
}

// ---------------------------------------------------------------------------
// p = softmax_j(u_j masked by v_mask), per batch (row-independent pairwise trick)
// ---------------------------------------------------------------------------
__global__ __launch_bounds__(256)
void k_pvec(const float* __restrict__ u, const int* __restrict__ v_mask,
            float* __restrict__ p) {
    __shared__ float red[256];
    const int b = blockIdx.x, t = threadIdx.x;
    const float* ub = u + (size_t)b * NV_;
    const int*   mb = v_mask + (size_t)b * NV_;
    float l[4];
    float mx = -3.4e38f;
#pragma unroll
    for (int j = 0; j < 4; ++j) {
        int jj = t + j * 256;
        float val = (mb[jj] == 0) ? NEGBIG : ub[jj];
        l[j] = val; mx = fmaxf(mx, val);
    }
    red[t] = mx; __syncthreads();
    for (int s = 128; s > 0; s >>= 1) { if (t < s) red[t] = fmaxf(red[t], red[t + s]); __syncthreads(); }
    mx = red[0]; __syncthreads();
    float sum = 0.0f;
#pragma unroll
    for (int j = 0; j < 4; ++j) { l[j] = __expf(l[j] - mx); sum += l[j]; }
    red[t] = sum; __syncthreads();
    for (int s = 128; s > 0; s >>= 1) { if (t < s) red[t] += red[t + s]; __syncthreads(); }
    float inv = 1.0f / red[0];
#pragma unroll
    for (int j = 0; j < 4; ++j) p[(size_t)b * NV_ + t + j * 256] = l[j] * inv;
}

// ---------------------------------------------------------------------------
// vbar[b,d] = sum_j p[b,j] * v[b,j,d]   (rank-1 result of the f-GEMM)
// ---------------------------------------------------------------------------
__global__ __launch_bounds__(256)
void k_vbar(const float* __restrict__ p, const float* __restrict__ v,
            float* __restrict__ vbar) {
    const int b = blockIdx.y;
    const int d = blockIdx.x * 256 + threadIdx.x;
    const float* pb = p + (size_t)b * NV_;
    const float* vb = v + (size_t)b * NV_ * D_;
    float acc = 0.0f;
    for (int j = 0; j < NV_; ++j) acc += pb[j] * vb[(size_t)j * D_ + d];
    vbar[(size_t)b * D_ + d] = acc;
}

// ---------------------------------------------------------------------------
// Broadcast vbar into h_new and z_new[:, D:2D]
// ---------------------------------------------------------------------------
__global__ __launch_bounds__(256)
void k_broadcast(const float* __restrict__ vbar, float* __restrict__ hout,
                 float* __restrict__ zout) {
    const int b = blockIdx.z, i = blockIdx.y;
    const int d = blockIdx.x * 256 + threadIdx.x;
    float val = vbar[(size_t)b * D_ + d];
    hout[((size_t)b * NV_ + i) * D_ + d] = val;
    zout[((size_t)b * NV_ + i) * (size_t)(2 * D_) + D_ + d] = val;
}

// ---------------------------------------------------------------------------
extern "C" void kernel_launch(void* const* d_in, const int* in_sizes, int n_in,
                              void* d_out, int out_size, void* d_ws, size_t ws_size,
                              hipStream_t stream) {
    (void)in_sizes; (void)n_in; (void)out_size; (void)ws_size;
    const float* v      = (const float*)d_in[0];
    const float* h      = (const float*)d_in[1];
    const float* z      = (const float*)d_in[2];
    const int*   v_mask = (const int*)d_in[3];
    const int*   h_mask = (const int*)d_in[4];
    const float* Wv_w   = (const float*)d_in[5];
    const float* Wv_b   = (const float*)d_in[6];
    const float* Wh_w   = (const float*)d_in[7];
    const float* qv_w   = (const float*)d_in[8];
    const float* qv_b   = (const float*)d_in[9];
    const float* kh_w   = (const float*)d_in[10];
    const float* Wz_w   = (const float*)d_in[11];
    const float* Wu_w   = (const float*)d_in[12];

    char* ws = (char*)d_ws;
    size_t off = 0;
    auto alloc = [&](size_t bytes) -> char* {
        char* p = ws + off;
        off = (off + bytes + 255) & ~(size_t)255;
        return p;
    };

    const size_t BND = (size_t)B_ * NV_ * D_;   // 16M elements
    const size_t DD  = (size_t)D_ * D_;

    unsigned short* vb     = (unsigned short*)alloc(BND * 2);
    unsigned short* hb     = (unsigned short*)alloc(BND * 2);
    unsigned short* zb     = (unsigned short*)alloc(BND * 4);        // [B,NV,2D]
    unsigned short* Wvb    = (unsigned short*)alloc(DD * 2);
    unsigned short* Whb    = (unsigned short*)alloc(DD * 2);
    unsigned short* qvb    = (unsigned short*)alloc(DD * 2);
    unsigned short* khb    = (unsigned short*)alloc(DD * 2);
    unsigned short* Wzb    = (unsigned short*)alloc(DD * 4);         // [D,2D]
    unsigned short* queryb = (unsigned short*)alloc(BND * 2);
    unsigned short* keyb   = (unsigned short*)alloc(BND * 2);
    unsigned short* hhb    = (unsigned short*)alloc(BND * 2);
    float*          s_pre  = (float*)alloc(BND * 4);
    float*          att    = (float*)alloc(BND * 4);                 // NH_==D_
    unsigned short* attp   = (unsigned short*)alloc(BND * 2);
    float*          ubuf   = (float*)alloc((size_t)B_ * NV_ * 4);
    float*          pbuf   = (float*)alloc((size_t)B_ * NV_ * 4);
    float*          vbar   = (float*)alloc((size_t)B_ * D_ * 4);

    auto cvt = [&](const float* s, unsigned short* d, size_t n) {
        k_f32_to_bf16<<<dim3(4096), dim3(256), 0, stream>>>(s, d, n);
    };
    cvt(v, vb, BND);
    cvt(h, hb, BND);
    cvt(z, zb, BND * 2);
    cvt(Wv_w, Wvb, DD);
    cvt(Wh_w, Whb, DD);
    cvt(qv_w, qvb, DD);
    cvt(kh_w, khb, DD);
    cvt(Wz_w, Wzb, DD * 2);

    const dim3 gg(4, 8, B_), gb(256);   // WG tile 128x256
    const long long sIO = (long long)NV_ * D_;

    // key = h @ kh^T                (bf16 out)
    k_gemm_bf16<0, false, false, true><<<gg, gb, 0, stream>>>(
        hb, sIO, khb, 0, nullptr, nullptr, keyb, sIO, NH_, D_, D_, D_, D_, D_);
    // query = v @ qv^T + qv_b       (bf16 out)
    k_gemm_bf16<0, true, false, true><<<gg, gb, 0, stream>>>(
        vb, sIO, qvb, 0, qv_b, nullptr, queryb, sIO, NV_, D_, D_, D_, D_, D_);
    // h_h = h @ Wh^T                (bf16 out)
    k_gemm_bf16<0, false, false, true><<<gg, gb, 0, stream>>>(
        hb, sIO, Whb, 0, nullptr, nullptr, hhb, sIO, NH_, D_, D_, D_, D_, D_);
    // s_pre = v @ Wv^T + Wv_b       (f32 out)
    k_gemm_bf16<0, true, false, false><<<gg, gb, 0, stream>>>(
        vb, sIO, Wvb, 0, Wv_b, s_pre, nullptr, sIO, NV_, D_, D_, D_, D_, D_);
    // s_pre += z @ Wz^T             (f32 accumulate, K=2048)
    k_gemm_bf16<0, false, true, false><<<gg, gb, 0, stream>>>(
        zb, (long long)NV_ * 2 * D_, Wzb, 0, nullptr, s_pre, nullptr, sIO,
        NV_, D_, 2 * D_, 2 * D_, 2 * D_, D_);
    // att = query @ key^T           (f32 out)
    k_gemm_bf16<0, false, false, false><<<gg, gb, 0, stream>>>(
        queryb, sIO, keyb, sIO, nullptr, att, nullptr, (long long)NV_ * NH_,
        NV_, NH_, D_, D_, D_, NH_);
    // probs = softmax(att, mask)    (bf16)
    k_softmax_att<<<dim3(NV_, B_), 256, 0, stream>>>(att, h_mask, attp);
    // s_pre += probs @ h_h          (NN, f32 accumulate)
    k_gemm_bf16<1, false, true, false><<<gg, gb, 0, stream>>>(
        attp, (long long)NV_ * NH_, hhb, sIO, nullptr, s_pre, nullptr, sIO,
        NV_, D_, NH_, NH_, D_, D_);

    float* hout = (float*)d_out;
    float* zout = hout + BND;
    // s = sigmoid(s_pre) -> z_new[:, :D]; u = s . Wu
    k_sigmoid_u<<<dim3(NV_, B_), 256, 0, stream>>>(s_pre, Wu_w, zout, ubuf);
    // p = softmax(u, v_mask)  (pairwise trick: row-independent)
    k_pvec<<<dim3(B_), 256, 0, stream>>>(ubuf, v_mask, pbuf);
    // vbar = p^T v
    k_vbar<<<dim3(D_ / 256, B_), 256, 0, stream>>>(pbuf, v, vbar);
    // h_new = broadcast(vbar); z_new[:, D:] = broadcast(vbar)
    k_broadcast<<<dim3(D_ / 256, NV_, B_), 256, 0, stream>>>(vbar, hout, zout);
}
